// CFM_90168543412306
// MI455X (gfx1250) — matrix operand order repaired and checked by reading, not compiled, not run on hardware
//
#include <hip/hip_runtime.h>
#include <hip/hip_bf16.h>
#include <math.h>

typedef __bf16 bf16;
typedef __attribute__((ext_vector_type(16))) __bf16 v16bf;
typedef __attribute__((ext_vector_type(8)))  float  v8f;
typedef __attribute__((ext_vector_type(4)))  unsigned int v4u;
typedef __attribute__((ext_vector_type(8)))  int v8i;
typedef __attribute__((ext_vector_type(4)))  int v4i;

static constexpr int BBATCH = 8, TT = 1024, MELD = 128, ENCD = 512, CD = 512, HHEADS = 8, FD = 2048;
static constexpr int DDIM = CD / HHEADS;  // 64

__device__ __forceinline__ float gelu_tanh(float x) {
  float x3 = x * x * x;
  return 0.5f * x * (1.f + tanhf(0.7978845608028654f * (x + 0.044715f * x3)));
}

__device__ __forceinline__ unsigned pack_bf16(float lo, float hi) {
  bf16 l = (bf16)lo, h = (bf16)hi;
  unsigned short ul, uh;
  __builtin_memcpy(&ul, &l, 2);
  __builtin_memcpy(&uh, &h, 2);
  return (unsigned)ul | ((unsigned)uh << 16);
}

// ---------------------------------------------------------------------------
// Tensor Data Mover: async 2D fp32 tile (rows x 32 cols, row stride rowStride)
// global -> LDS, +1 DWORD pad per 32 DWORDs (LDS row pitch = 33 floats).
// D# layout per CDNA5 ISA.
// ---------------------------------------------------------------------------
__device__ __forceinline__ void tdm_load_tile_f32(
    unsigned lds_off, const float* gsrc, int rows, int rowStride)
{
  unsigned long long ga = (unsigned long long)(uintptr_t)gsrc;
  v4u g0;
  g0[0] = 1u;                                       // count=1, user descriptor
  g0[1] = lds_off;                                  // lds_addr (bytes)
  g0[2] = (unsigned)ga;                             // global_addr[31:0]
  g0[3] = (unsigned)(ga >> 32) | 0x80000000u;       // global_addr[56:32] | type=2
  v8i g1;
  g1[0] = (int)((2u << 16) | (1u << 20) | (4u << 22)); // data_size=4B, pad_enable, pad_interval=32 DW
  g1[1] = (int)(((unsigned)TT & 0xFFFFu) << 16);       // tensor_dim0 low16  @ [63:48]
  g1[2] = (int)((((unsigned)TT) >> 16) | (((unsigned)rows & 0xFFFFu) << 16)); // dim0 hi | dim1 lo
  g1[3] = (int)((((unsigned)rows) >> 16) | (32u << 16));                      // dim1 hi | tile_dim0=32
  g1[4] = rows;                                     // tile_dim1 (tile_dim2 = 0)
  g1[5] = rowStride;                                // tensor_dim0_stride low 32
  g1[6] = 0;
  g1[7] = 0;
  v4i z4 = {0, 0, 0, 0};
#if __has_include(<hip/amd_detail/amd_gfx1250_TDM.h>)
  v8i z8 = {0, 0, 0, 0, 0, 0, 0, 0};
  __builtin_amdgcn_tensor_load_to_lds(g0, g1, z4, z4, z8, 0);
#else
  __builtin_amdgcn_tensor_load_to_lds(g0, g1, z4, z4, 0);
#endif
}

// ---------------------------------------------------------------------------
// Generic WMMA GEMM:  Out[b,m,n] = act( sum_k A[b,m,k]*B[b,k,n] + bias[m] ) + addSrc[b,m,n]
// conv3: B operand is implicit K=3 SAME conv (I = Kd/3 rows, row stride sBk, time length N).
// ---------------------------------------------------------------------------
#define BM 128
#define BN 128
#define BKK 32

__global__ __launch_bounds__(256) void gemm_kernel(
    const float* __restrict__ A, long sAm, long sAk, long sAb,
    const float* __restrict__ Bp, long sBk, long sBn, long sBb,
    const float* __restrict__ bias,
    const float* __restrict__ addSrc, long sAddB,
    float* __restrict__ Out, long sOutB,
    int M, int N, int Kd, int act, int conv3)
{
  __shared__ bf16 As[2][BM][BKK + 2];
  __shared__ bf16 Bs[2][BN][BKK + 2];
  const int tid = threadIdx.x;
  const int b   = blockIdx.z;
  const int m0  = blockIdx.y * BM;
  const int n0  = blockIdx.x * BN;
  const float* Ab = A + (size_t)b * sAb;
  const float* Bb = Bp + (size_t)b * sBb;

  const int w = tid >> 5, lane = tid & 31;
  const int wm = (w & 1) * 64;
  const int wn = (w >> 1) * 32;
  const int row  = lane & 15;
  const int half = lane >> 4;
  const int kgrpA  = half * 8;
  const int khalfB = half * 16;

  v8f acc[4][2];
  #pragma unroll
  for (int i = 0; i < 4; ++i)
    #pragma unroll
    for (int j = 0; j < 2; ++j)
      #pragma unroll
      for (int r = 0; r < 8; ++r) acc[i][j][r] = 0.f;

  auto stage = [&](int buf, int k0) {
    // ---- A tile: BM x BKK ----
    if (sAk == 1) {
      #pragma unroll
      for (int q = tid; q < BM * (BKK / 4); q += 256) {
        int m = q >> 3, kq = (q & 7) * 4;
        const float4 f = *reinterpret_cast<const float4*>(
            Ab + (size_t)(m0 + m) * sAm + (size_t)(k0 + kq));
        unsigned* dst = (unsigned*)&As[buf][m][kq];
        dst[0] = pack_bf16(f.x, f.y);
        dst[1] = pack_bf16(f.z, f.w);
      }
    } else {
      #pragma unroll
      for (int e = tid; e < BM * BKK; e += 256) {
        int m = e >> 5, k = e & 31;
        As[buf][m][k] = (bf16)Ab[(size_t)(m0 + m) * sAm + (size_t)(k0 + k) * sAk];
      }
    }
    // ---- B tile: BKK x BN, transposed into Bs[n][k] ----
    if (conv3) {
      #pragma unroll
      for (int e = tid; e < BKK * BN; e += 256) {
        int k = e >> 7, n = e & 127;
        int kk = k0 + k;
        int i  = kk / 3;
        int dk = kk - 3 * i - 1;
        int t  = n0 + n + dk;
        float vv = (t >= 0 && t < N) ? Bb[(size_t)i * sBk + t] : 0.f;
        Bs[buf][n][k] = (bf16)vv;
      }
    } else if (sBn == 1) {
      #pragma unroll
      for (int q = tid; q < BKK * (BN / 4); q += 256) {
        int k = q >> 5, nq = (q & 31) * 4;
        const float4 f = *reinterpret_cast<const float4*>(
            Bb + (size_t)(k0 + k) * sBk + (size_t)(n0 + nq));
        Bs[buf][nq + 0][k] = (bf16)f.x;
        Bs[buf][nq + 1][k] = (bf16)f.y;
        Bs[buf][nq + 2][k] = (bf16)f.z;
        Bs[buf][nq + 3][k] = (bf16)f.w;
      }
    } else {
      #pragma unroll
      for (int e = tid; e < BKK * BN; e += 256) {
        int k = e >> 7, n = e & 127;
        Bs[buf][n][k] = (bf16)Bb[(size_t)(k0 + k) * sBk + (size_t)(n0 + n) * sBn];
      }
    }
  };

  const int nk = Kd / BKK;
  stage(0, 0);
  __syncthreads();
  for (int kt = 0; kt < nk; ++kt) {
    const int buf = kt & 1;
    if (kt + 1 < nk) stage(buf ^ 1, (kt + 1) * BKK);

    v16bf afr[4];
    #pragma unroll
    for (int ms = 0; ms < 4; ++ms) {
      const bf16* ap = &As[buf][wm + ms * 16 + row][0];
      #pragma unroll
      for (int i = 0; i < 4; ++i) {
        int k = kgrpA + 2 * i;
        afr[ms][2 * i]     = ap[k];
        afr[ms][2 * i + 1] = ap[k + 1];
      }
      #pragma unroll
      for (int i = 0; i < 4; ++i) {
        int k = 16 + kgrpA + 2 * i;
        afr[ms][8 + 2 * i]     = ap[k];
        afr[ms][8 + 2 * i + 1] = ap[k + 1];
      }
    }
    v16bf bfr[2];
    #pragma unroll
    for (int ns = 0; ns < 2; ++ns) {
      const bf16* bp = &Bs[buf][wn + ns * 16 + row][0];
      #pragma unroll
      for (int i = 0; i < 8; ++i) {
        int k = (i < 4) ? (kgrpA + 2 * i) : (16 + kgrpA + 2 * (i - 4));
        bfr[ns][2 * i]     = bp[k];
        bfr[ns][2 * i + 1] = bp[k + 1];
      }
    }
    #pragma unroll
    for (int ms = 0; ms < 4; ++ms)
      #pragma unroll
      for (int ns = 0; ns < 2; ++ns)
        acc[ms][ns] = __builtin_amdgcn_wmma_f32_16x16x32_bf16(
            false, afr[ms], false, bfr[ns], (short)0, acc[ms][ns], false, false);
    __syncthreads();
  }

  #pragma unroll
  for (int ms = 0; ms < 4; ++ms) {
    #pragma unroll
    for (int ns = 0; ns < 2; ++ns) {
      #pragma unroll
      for (int r = 0; r < 8; ++r) {
        int m = m0 + wm + ms * 16 + r + 8 * half;
        int n = n0 + wn + ns * 16 + row;
        float val = acc[ms][ns][r];
        if (bias) val += bias[m];
        if (act == 1) val = gelu_tanh(val);
        else if (act == 2) val = fmaxf(val, 0.f);
        if (addSrc) val += addSrc[(size_t)b * sAddB + (size_t)m * N + n];
        Out[(size_t)b * sOutB + (size_t)m * N + n] = val;
      }
    }
  }
}

// ---------------------------------------------------------------------------
// Fused flash attention. q,k,v,out: (B, C, T).  grid (T/128, H, B), 8 waves.
// K/V tiles (64 x 32 fp32, LDS pitch 33) staged once per block by wave 0 via
// TDM (double-buffered, s_wait_tensorcnt), shared by all 8 waves.
// ---------------------------------------------------------------------------
__global__ __launch_bounds__(256) void attn_kernel(
    const float* __restrict__ q, const float* __restrict__ k,
    const float* __restrict__ v, const int* __restrict__ lengths,
    float* __restrict__ outp)
{
  __shared__ float Ks[2][64][33];
  __shared__ float Vs[2][64][33];
  __shared__ bf16 Ps[8][16][34];
  const int b = blockIdx.z, h = blockIdx.y;
  const int w = threadIdx.x >> 5, lane = threadIdx.x & 31;
  const int q0 = blockIdx.x * 128 + w * 16;
  const int row = lane & 15, half = lane >> 4;
  const int kgrpA = half * 8, khalfB = half * 16;
  const int len = lengths[b];
  const float* qb = q + ((size_t)b * CD + h * DDIM) * TT;
  const float* kb = k + ((size_t)b * CD + h * DDIM) * TT;
  const float* vb = v + ((size_t)b * CD + h * DDIM) * TT;
  const int tq = q0 + row;

  v16bf qf[2];
  #pragma unroll
  for (int c = 0; c < 2; ++c) {
    #pragma unroll
    for (int i = 0; i < 8; ++i) {
      int dk = (i < 4) ? (kgrpA + 2 * i) : (16 + kgrpA + 2 * (i - 4));
      int d = c * 32 + dk;
      qf[c][2 * i]     = (bf16)qb[(size_t)d * TT + tq];
      qf[c][2 * i + 1] = (bf16)qb[(size_t)(d + 1) * TT + tq];
    }
  }

  v8f oacc[4];
  #pragma unroll
  for (int ds = 0; ds < 4; ++ds)
    #pragma unroll
    for (int r = 0; r < 8; ++r) oacc[ds][r] = 0.f;
  float mrow[8], lrow[8];
  #pragma unroll
  for (int r = 0; r < 8; ++r) { mrow[r] = -3.0e38f; lrow[r] = 0.f; }

  const int nk = TT / 32;
  if (w == 0) {
    tdm_load_tile_f32((unsigned)(uintptr_t)&Ks[0][0][0], kb, 64, TT);
    tdm_load_tile_f32((unsigned)(uintptr_t)&Vs[0][0][0], vb, 64, TT);
    __builtin_amdgcn_s_wait_tensorcnt(0);
  }
  __syncthreads();

  for (int kt = 0; kt < nk; ++kt) {
    const int buf = kt & 1;
    const int tk0 = kt * 32;
    if (w == 0 && kt + 1 < nk) {
      tdm_load_tile_f32((unsigned)(uintptr_t)&Ks[buf ^ 1][0][0], kb + (kt + 1) * 32, 64, TT);
      tdm_load_tile_f32((unsigned)(uintptr_t)&Vs[buf ^ 1][0][0], vb + (kt + 1) * 32, 64, TT);
    }

    v8f sac[2];
    #pragma unroll
    for (int ns = 0; ns < 2; ++ns)
      #pragma unroll
      for (int r = 0; r < 8; ++r) sac[ns][r] = 0.f;

    #pragma unroll
    for (int ns = 0; ns < 2; ++ns) {
      const int tkl = ns * 16 + row;
      #pragma unroll
      for (int c = 0; c < 2; ++c) {
        v16bf bfr;
        #pragma unroll
        for (int i = 0; i < 8; ++i) {
          int d = c * 32 + ((i < 4) ? (kgrpA + 2 * i) : (16 + kgrpA + 2 * (i - 4)));
          bfr[2 * i]     = (bf16)Ks[buf][d][tkl];
          bfr[2 * i + 1] = (bf16)Ks[buf][d + 1][tkl];
        }
        sac[ns] = __builtin_amdgcn_wmma_f32_16x16x32_bf16(
            false, qf[c], false, bfr, (short)0, sac[ns], false, false);
      }
    }

    float p0s[8], p1s[8];
    #pragma unroll
    for (int r = 0; r < 8; ++r) {
      float s0 = sac[0][r] * 0.125f;
      float s1 = sac[1][r] * 0.125f;
      if (tk0 + row >= len)      s0 = -1.0e9f;
      if (tk0 + 16 + row >= len) s1 = -1.0e9f;
      float mx = fmaxf(s0, s1);
      #pragma unroll
      for (int off = 8; off; off >>= 1) mx = fmaxf(mx, __shfl_xor(mx, off, 32));
      float mnew  = fmaxf(mrow[r], mx);
      float alpha = __expf(mrow[r] - mnew);
      float p0 = __expf(s0 - mnew);
      float p1 = __expf(s1 - mnew);
      float rs = p0 + p1;
      #pragma unroll
      for (int off = 8; off; off >>= 1) rs += __shfl_xor(rs, off, 32);
      lrow[r] = lrow[r] * alpha + rs;
      mrow[r] = mnew;
      #pragma unroll
      for (int ds = 0; ds < 4; ++ds) oacc[ds][r] *= alpha;
      p0s[r] = p0; p1s[r] = p1;
    }

    #pragma unroll
    for (int r = 0; r < 8; ++r) {
      int m = r + 8 * half;
      Ps[w][m][row]      = (bf16)p0s[r];
      Ps[w][m][16 + row] = (bf16)p1s[r];
    }
    asm volatile("s_wait_dscnt 0x0" ::: "memory");  // wave-internal LDS visibility

    v16bf pf;
    {
      const bf16* pp = &Ps[w][row][0];
      #pragma unroll
      for (int i = 0; i < 4; ++i) { int kk = kgrpA + 2 * i;      pf[2*i] = pp[kk]; pf[2*i+1] = pp[kk+1]; }
      #pragma unroll
      for (int i = 0; i < 4; ++i) { int kk = 16 + kgrpA + 2 * i; pf[8+2*i] = pp[kk]; pf[8+2*i+1] = pp[kk+1]; }
    }

    #pragma unroll
    for (int ds = 0; ds < 4; ++ds) {
      v16bf vfr;
      const int d = ds * 16 + row;
      #pragma unroll
      for (int i = 0; i < 8; ++i) {
        int kk = (i < 4) ? (kgrpA + 2 * i) : (16 + kgrpA + 2 * (i - 4));
        vfr[2 * i]     = (bf16)Vs[buf][d][kk];
        vfr[2 * i + 1] = (bf16)Vs[buf][d][kk + 1];
      }
      oacc[ds] = __builtin_amdgcn_wmma_f32_16x16x32_bf16(
          false, pf, false, vfr, (short)0, oacc[ds], false, false);
    }

    __syncthreads();                                   // all waves done with buf
    if (w == 0 && kt + 1 < nk) __builtin_amdgcn_s_wait_tensorcnt(0);
    __syncthreads();                                   // next buffer visible
  }

  #pragma unroll
  for (int ds = 0; ds < 4; ++ds) {
    #pragma unroll
    for (int r = 0; r < 8; ++r) {
      int m = q0 + r + 8 * half;
      int d = ds * 16 + row;
      outp[((size_t)b * CD + h * DDIM + d) * TT + m] = oacc[ds][r] / lrow[r];
    }
  }
}

// --------------------------- LayerNorm / CLN -------------------------------
__global__ void ln_kernel(float* __restrict__ y,
                          const float* __restrict__ g, const float* __restrict__ bvec,
                          const float* __restrict__ gamma, const float* __restrict__ beta,
                          const int* __restrict__ lengths, int applyMask)
{
  int idx = blockIdx.x * blockDim.x + threadIdx.x;
  if (idx >= BBATCH * TT) return;
  int b = idx / TT, t = idx - b * TT;
  float* col = y + (size_t)b * CD * TT + t;
  float s = 0.f, ss = 0.f;
  for (int c = 0; c < CD; ++c) { float vv = col[(size_t)c * TT]; s += vv; ss += vv * vv; }
  float mu = s / CD;
  float var = ss / CD - mu * mu;
  float rinv = rsqrtf(var + 1e-5f);
  float maskf = 1.f;
  if (applyMask) maskf = (t < lengths[b]) ? 1.f : 0.f;
  for (int c = 0; c < CD; ++c) {
    float hh = (col[(size_t)c * TT] - mu) * rinv;
    if (gamma) {
      size_t gi = ((size_t)b * CD + c) * TT + t;
      hh = hh * (1.f + gamma[gi]) + beta[gi];
    } else {
      hh = hh * g[c] + bvec[c];
    }
    col[(size_t)c * TT] = hh * maskf;
  }
}

// --------------------------- Sinkhorn helpers ------------------------------
__global__ void colnorm_kernel(const float* __restrict__ x, float* __restrict__ outn, int Mrows)
{
  int idx = blockIdx.x * blockDim.x + threadIdx.x;
  if (idx >= BBATCH * TT) return;
  int b = idx / TT, t = idx - b * TT;
  const float* p = x + (size_t)b * Mrows * TT + t;
  float s = 0.f;
  for (int m = 0; m < Mrows; ++m) { float vv = p[(size_t)m * TT]; s += vv * vv; }
  outn[idx] = s;
}

__global__ void mr_kernel(float* __restrict__ G, float* __restrict__ MrT,
                          const float* __restrict__ xsq, const float* __restrict__ ysq)
{
  __shared__ float tile[32][33];
  int b = blockIdx.z;
  int t0 = blockIdx.y * 32, s0 = blockIdx.x * 32;
  for (int r = threadIdx.y; r < 32; r += 8) {
    int t = t0 + r, s = s0 + threadIdx.x;
    size_t gi = ((size_t)b * TT + t) * TT + s;
    float d = xsq[b * TT + t] + ysq[b * TT + s] - 2.f * G[gi];
    d = fmaxf(d, 0.f) * (-100.f);   // -d / EPS, EPS = 0.01
    G[gi] = d;
    tile[r][threadIdx.x] = d;
  }
  __syncthreads();
  for (int r = threadIdx.y; r < 32; r += 8) {
    int s = s0 + r, t = t0 + threadIdx.x;
    MrT[((size_t)b * TT + s) * TT + t] = tile[threadIdx.x][r];
  }
}

// out[row] = logc - LSE_j( M[row, j] + wv[b, j] );  wave per row, float4 lanes.
__global__ __launch_bounds__(256) void lse_kernel(
    const float* __restrict__ M, const float* __restrict__ wv,
    float* __restrict__ outv, float logc)
{
  int rowid = blockIdx.x * 8 + (threadIdx.x >> 5);
  int lane = threadIdx.x & 31;
  int b = rowid / TT;
  const float* mr = M + (size_t)rowid * TT;
  const float* wb = wv + (size_t)b * TT;
  float mx = -3.0e38f, acc = 0.f;
  for (int j = lane * 4; j < TT; j += 128) {
    const float4 a  = *reinterpret_cast<const float4*>(mr + j);
    const float4 w4 = *reinterpret_cast<const float4*>(wb + j);
    float vals[4] = {a.x + w4.x, a.y + w4.y, a.z + w4.z, a.w + w4.w};
    #pragma unroll
    for (int e = 0; e < 4; ++e) {
      float val = vals[e];
      if (val > mx) { acc = acc * __expf(mx - val) + 1.f; mx = val; }
      else acc += __expf(val - mx);
    }
  }
  for (int off = 16; off; off >>= 1) {
    float omx = __shfl_xor(mx, off, 32);
    float oac = __shfl_xor(acc, off, 32);
    float nm = fmaxf(mx, omx);
    acc = acc * __expf(mx - nm) + oac * __expf(omx - nm);
    mx = nm;
  }
  if (lane == 0) outv[rowid] = logc - (mx + __logf(acc));
}

// otT[b, s, t] = exp(MrT[b,s,t] + u[b,t] + v[b,s]);  4 elements per thread.
__global__ void ot_kernel(const float* __restrict__ MrT, const float* __restrict__ u,
                          const float* __restrict__ v, float* __restrict__ otT)
{
  size_t i4 = ((size_t)blockIdx.x * blockDim.x + threadIdx.x) * 4;
  size_t total = (size_t)BBATCH * TT * TT;
  if (i4 >= total) return;
  size_t per = (size_t)TT * TT;
  int b = (int)(i4 / per);
  size_t r = i4 - (size_t)b * per;
  int s = (int)(r / TT), t = (int)(r - (size_t)s * TT);
  const float4 m4 = *reinterpret_cast<const float4*>(MrT + i4);
  const float vs = v[b * TT + s];
  float4 o;
  o.x = __expf(m4.x + u[b * TT + t + 0] + vs);
  o.y = __expf(m4.y + u[b * TT + t + 1] + vs);
  o.z = __expf(m4.z + u[b * TT + t + 2] + vs);
  o.w = __expf(m4.w + u[b * TT + t + 3] + vs);
  *reinterpret_cast<float4*>(otT + i4) = o;
}

// --------------------------- small elementwise -----------------------------
__global__ void zero_kernel(float* p, int n)
{ int i = blockIdx.x * blockDim.x + threadIdx.x; if (i < n) p[i] = 0.f; }

__global__ void noised_kernel(const float* __restrict__ mels, const float* __restrict__ transported,
                              const float* __restrict__ times, float* __restrict__ noised)
{
  int idx = blockIdx.x * blockDim.x + threadIdx.x;
  if (idx >= BBATCH * MELD * TT) return;
  int b = idx / (MELD * TT);
  float tt = times[b];
  noised[idx] = tt * mels[idx] + (1.f - tt) * transported[idx];
}

__global__ void timeemb_kernel(const float* __restrict__ times, float* __restrict__ ebuf)
{
  int idx = blockIdx.x * blockDim.x + threadIdx.x;
  if (idx >= BBATCH * CD) return;
  int b = idx / CD, c = idx - b * CD;
  const int halfc = CD / 2;
  float lf = -logf(10000.f) / (float)(halfc - 1);
  int cc = (c < halfc) ? c : c - halfc;
  float ang = times[b] * __expf((float)cc * lf);
  ebuf[idx] = (c < halfc) ? __sinf(ang) : __cosf(ang);
}

__global__ void step1_kernel(const float* __restrict__ w1, const float* __restrict__ b1,
                             const float* __restrict__ ebuf, float* __restrict__ s1)
{
  int idx = blockIdx.x * blockDim.x + threadIdx.x;
  if (idx >= BBATCH * FD) return;
  int b = idx / FD, f = idx - b * FD;
  float s = b1[f];
  for (int c = 0; c < CD; ++c) s += w1[(size_t)f * CD + c] * ebuf[b * CD + c];
  s1[idx] = gelu_tanh(s);
}

__global__ void step2_kernel(const float* __restrict__ w2, const float* __restrict__ b2,
                             const float* __restrict__ s1, float* __restrict__ sbuf)
{
  int idx = blockIdx.x * blockDim.x + threadIdx.x;
  if (idx >= BBATCH * CD) return;
  int b = idx / CD, c = idx - b * CD;
  float s = b2[c];
  for (int f = 0; f < FD; ++f) s += w2[(size_t)c * FD + f] * s1[b * FD + f];
  sbuf[idx] = s;
}

__global__ void condadd_kernel(float* __restrict__ cond, const float* __restrict__ sing2,
                               const float* __restrict__ sbuf)
{
  int idx = blockIdx.x * blockDim.x + threadIdx.x;
  if (idx >= BBATCH * CD * TT) return;
  int b = idx / (CD * TT);
  int c = (idx / TT) % CD;
  cond[idx] += sing2[idx] + sbuf[b * CD + c];
}

__global__ void final_kernel(const float* __restrict__ mels, const float* __restrict__ noises,
                             const float* __restrict__ noised, const float* __restrict__ netout,
                             const float* __restrict__ times, float* __restrict__ out)
{
  int idx = blockIdx.x * blockDim.x + threadIdx.x;
  const size_t NN = (size_t)BBATCH * MELD * TT;
  if (idx >= (int)NN) return;
  int b = idx / (MELD * TT);
  float tt = times[b];
  float m = mels[idx], nz = noises[idx], nm = noised[idx], pf = netout[idx];
  out[idx]            = m - nz;
  out[NN + idx]       = m - nm;
  out[2 * NN + idx]   = pf;
  out[3 * NN + idx]   = nz;
  out[4 * NN + idx]   = nm - pf * fmaxf(tt, 1e-7f);
}

// ===========================================================================
extern "C" void kernel_launch(void* const* d_in, const int* in_sizes, int n_in,
                              void* d_out, int out_size, void* d_ws, size_t ws_size,
                              hipStream_t stream) {
  (void)in_sizes; (void)n_in; (void)out_size; (void)ws_size;
  auto F32 = [&](int i) { return (const float*)d_in[i]; };

  const float* encodings = F32(0);
  const float* singers   = F32(1);
  const float* mels      = F32(2);
  const float* noises    = F32(3);
  const float* times     = F32(4);
  const int*   lengths   = (const int*)d_in[5];

  struct Blk { const float *bk,*bo,*bq,*bv,*cw1b,*cw1g,*cw2b,*cw2g,*fb1,*fb2,*fw1,*fw2,*l1b,*l1g,*l2b,*l2g,*wk,*wo,*wq,*wv; };
  Blk blk[4];
  int pi = 6;
  blk[0].bk=F32(pi++); blk[0].bo=F32(pi++); blk[0].bq=F32(pi++); blk[0].bv=F32(pi++);
  blk[0].cw1b=F32(pi++); blk[0].cw1g=F32(pi++); blk[0].cw2b=F32(pi++); blk[0].cw2g=F32(pi++);
  blk[0].fb1=F32(pi++); blk[0].fb2=F32(pi++); blk[0].fw1=F32(pi++); blk[0].fw2=F32(pi++);
  blk[0].l1b=F32(pi++); blk[0].l1g=F32(pi++); blk[0].l2b=F32(pi++); blk[0].l2g=F32(pi++);
  blk[0].wk=F32(pi++); blk[0].wo=F32(pi++); blk[0].wq=F32(pi++); blk[0].wv=F32(pi++);
  for (int j = 1; j < 4; ++j) {
    blk[j].bk=F32(pi++); blk[j].bo=F32(pi++); blk[j].bq=F32(pi++); blk[j].bv=F32(pi++);
    blk[j].fb1=F32(pi++); blk[j].fb2=F32(pi++); blk[j].fw1=F32(pi++); blk[j].fw2=F32(pi++);
    blk[j].l1b=F32(pi++); blk[j].l1g=F32(pi++); blk[j].l2b=F32(pi++); blk[j].l2g=F32(pi++);
    blk[j].wk=F32(pi++); blk[j].wo=F32(pi++); blk[j].wq=F32(pi++); blk[j].wv=F32(pi++);
    blk[j].cw1b = blk[j].cw1g = blk[j].cw2b = blk[j].cw2g = nullptr;
  }
  const float *enc_b1=F32(pi++), *enc_b2=F32(pi++), *enc_w1=F32(pi++), *enc_w2=F32(pi++);
  const float *prenet_b=F32(pi++), *prenet_w=F32(pi++);
  const float *proj_b1=F32(pi++), *proj_b2=F32(pi++), *proj_w1=F32(pi++), *proj_w2=F32(pi++);
  const float *sing_b1=F32(pi++), *sing_b2=F32(pi++), *sing_w1=F32(pi++), *sing_w2=F32(pi++);
  const float *step_b1=F32(pi++), *step_b2=F32(pi++), *step_w1=F32(pi++), *step_w2=F32(pi++);

  char* base = (char*)d_ws;
  size_t off = 0;
  auto alloc = [&](size_t bytes) -> void* {
    void* p = base + off;
    off += (bytes + 255) & ~(size_t)255;
    return p;
  };
  const size_t TT2 = (size_t)TT * TT;
  float* Mr  = (float*)alloc((size_t)BBATCH * TT2 * 4);
  float* MrT = (float*)alloc((size_t)BBATCH * TT2 * 4);
  float* otT = (float*)alloc((size_t)BBATCH * TT2 * 4);
  float* uvec = (float*)alloc((size_t)BBATCH * TT * 4);
  float* vvec = (float*)alloc((size_t)BBATCH * TT * 4);
  float* xsq  = (float*)alloc((size_t)BBATCH * TT * 4);
  float* ysq  = (float*)alloc((size_t)BBATCH * TT * 4);
  float* transported = (float*)alloc((size_t)BBATCH * MELD * TT * 4);
  float* noised      = (float*)alloc((size_t)BBATCH * MELD * TT * 4);
  float* xb   = (float*)alloc((size_t)BBATCH * CD * TT * 4);
  float* yb   = (float*)alloc((size_t)BBATCH * CD * TT * 4);
  float* cond = (float*)alloc((size_t)BBATCH * CD * TT * 4);
  float* qbuf = (float*)alloc((size_t)BBATCH * CD * TT * 4);
  float* kbuf = (float*)alloc((size_t)BBATCH * CD * TT * 4);
  float* vbuf = (float*)alloc((size_t)BBATCH * CD * TT * 4);
  float* abuf = (float*)alloc((size_t)BBATCH * CD * TT * 4);
  float* sing2  = (float*)alloc((size_t)BBATCH * CD * TT * 4);
  float* netout = (float*)alloc((size_t)BBATCH * MELD * TT * 4);
  float* ebuf = (float*)alloc((size_t)BBATCH * CD * 4);
  float* s1buf = (float*)alloc((size_t)BBATCH * FD * 4);
  float* sbuf = (float*)alloc((size_t)BBATCH * CD * 4);
  float* tmpF   = Mr;                                   // reuse 64 MB after sinkhorn
  float* gammab = otT;                                  // reuse otT region for CLN
  float* betab  = otT + (size_t)BBATCH * CD * TT;
  float* pbuf   = qbuf;

  auto gemm = [&](const float* A, long sAm, long sAk, long sAb,
                  const float* Bp, long sBk, long sBn, long sBb,
                  const float* bias, const float* addSrc, long sAddB,
                  float* Out, long sOutB, int M, int N, int Kd, int act, int conv3) {
    dim3 g(N / BN, M / BM, BBATCH);
    gemm_kernel<<<g, 256, 0, stream>>>(A, sAm, sAk, sAb, Bp, sBk, sBn, sBb,
                                       bias, addSrc, sAddB, Out, sOutB, M, N, Kd, act, conv3);
  };

  // ===== 1) OT plan =====
  colnorm_kernel<<<(BBATCH * TT + 255) / 256, 256, 0, stream>>>(noises, xsq, MELD);
  colnorm_kernel<<<(BBATCH * TT + 255) / 256, 256, 0, stream>>>(mels,   ysq, MELD);
  gemm(noises, 1, TT, (long)MELD * TT, mels, TT, 1, (long)MELD * TT,
       nullptr, nullptr, 0, Mr, (long)TT2, TT, TT, MELD, 0, 0);
  mr_kernel<<<dim3(TT / 32, TT / 32, BBATCH), dim3(32, 8), 0, stream>>>(Mr, MrT, xsq, ysq);
  zero_kernel<<<(BBATCH * TT + 255) / 256, 256, 0, stream>>>(uvec, BBATCH * TT);
  zero_kernel<<<(BBATCH * TT + 255) / 256, 256, 0, stream>>>(vvec, BBATCH * TT);
  const float logc = -6.9314718056f;   // -log(1024)
  for (int it = 0; it < 100; ++it) {
    lse_kernel<<<BBATCH * TT / 8, 256, 0, stream>>>(Mr,  vvec, uvec, logc);
    lse_kernel<<<BBATCH * TT / 8, 256, 0, stream>>>(MrT, uvec, vvec, logc);
  }
  ot_kernel<<<(int)((size_t)BBATCH * TT2 / (256 * 4)), 256, 0, stream>>>(MrT, uvec, vvec, otT);
  gemm(noises, TT, 1, (long)MELD * TT, otT, TT, 1, (long)TT2,
       nullptr, nullptr, 0, transported, (long)MELD * TT, MELD, TT, TT, 0, 0);
  noised_kernel<<<(BBATCH * MELD * TT + 255) / 256, 256, 0, stream>>>(mels, transported, times, noised);

  // ===== 2) network =====
  gemm(prenet_w, MELD, 1, 0, noised, TT, 1, (long)MELD * TT,
       prenet_b, nullptr, 0, xb, (long)CD * TT, CD, TT, MELD, 1, 0);
  gemm(enc_w1, MELD, 1, 0, encodings, TT, 1, (long)MELD * TT,
       enc_b1, nullptr, 0, tmpF, (long)FD * TT, FD, TT, MELD, 1, 0);
  gemm(enc_w2, FD, 1, 0, tmpF, TT, 1, (long)FD * TT,
       enc_b2, nullptr, 0, cond, (long)CD * TT, CD, TT, FD, 0, 0);
  gemm(sing_w1, ENCD, 1, 0, singers, TT, 1, (long)ENCD * TT,
       sing_b1, nullptr, 0, tmpF, (long)FD * TT, FD, TT, ENCD, 1, 0);
  gemm(sing_w2, FD, 1, 0, tmpF, TT, 1, (long)FD * TT,
       sing_b2, nullptr, 0, sing2, (long)CD * TT, CD, TT, FD, 0, 0);
  timeemb_kernel<<<(BBATCH * CD + 255) / 256, 256, 0, stream>>>(times, ebuf);
  step1_kernel<<<(BBATCH * FD + 255) / 256, 256, 0, stream>>>(step_w1, step_b1, ebuf, s1buf);
  step2_kernel<<<(BBATCH * CD + 255) / 256, 256, 0, stream>>>(step_w2, step_b2, s1buf, sbuf);
  condadd_kernel<<<(BBATCH * CD * TT + 255) / 256, 256, 0, stream>>>(cond, sing2, sbuf);

  float* xin = xb;
  float* yout = yb;
  for (int i = 0; i < 4; ++i) {
    const Blk& bp = blk[i];
    gemm(bp.wq, CD, 1, 0, xin, TT, 1, (long)CD * TT, bp.bq, nullptr, 0, qbuf, (long)CD * TT, CD, TT, CD, 0, 0);
    gemm(bp.wk, CD, 1, 0, xin, TT, 1, (long)CD * TT, bp.bk, nullptr, 0, kbuf, (long)CD * TT, CD, TT, CD, 0, 0);
    gemm(bp.wv, CD, 1, 0, xin, TT, 1, (long)CD * TT, bp.bv, nullptr, 0, vbuf, (long)CD * TT, CD, TT, CD, 0, 0);
    attn_kernel<<<dim3(TT / 128, HHEADS, BBATCH), 256, 0, stream>>>(qbuf, kbuf, vbuf, lengths, abuf);
    gemm(bp.wo, CD, 1, 0, abuf, TT, 1, (long)CD * TT, bp.bo, xin, (long)CD * TT,
         yout, (long)CD * TT, CD, TT, CD, 0, 0);
    if (i == 0) {
      gemm(bp.cw1g, CD, 1, 0, cond, TT, 1, (long)CD * TT, nullptr, nullptr, 0, gammab, (long)CD * TT, CD, TT, CD, 0, 0);
      gemm(bp.cw1b, CD, 1, 0, cond, TT, 1, (long)CD * TT, nullptr, nullptr, 0, betab,  (long)CD * TT, CD, TT, CD, 0, 0);
      ln_kernel<<<(BBATCH * TT + 255) / 256, 256, 0, stream>>>(yout, nullptr, nullptr, gammab, betab, lengths, 0);
    } else {
      ln_kernel<<<(BBATCH * TT + 255) / 256, 256, 0, stream>>>(yout, bp.l1g, bp.l1b, nullptr, nullptr, lengths, 0);
    }
    gemm(bp.fw1, (long)CD * 3, 1, 0, yout, TT, 1, (long)CD * TT,
         bp.fb1, nullptr, 0, tmpF, (long)FD * TT, FD, TT, CD * 3, 2, 1);
    gemm(bp.fw2, (long)FD * 3, 1, 0, tmpF, TT, 1, (long)FD * TT,
         bp.fb2, yout, (long)CD * TT, yout, (long)CD * TT, CD, TT, FD * 3, 0, 1);
    if (i == 0) {
      gemm(bp.cw2g, CD, 1, 0, cond, TT, 1, (long)CD * TT, nullptr, nullptr, 0, gammab, (long)CD * TT, CD, TT, CD, 0, 0);
      gemm(bp.cw2b, CD, 1, 0, cond, TT, 1, (long)CD * TT, nullptr, nullptr, 0, betab,  (long)CD * TT, CD, TT, CD, 0, 0);
      ln_kernel<<<(BBATCH * TT + 255) / 256, 256, 0, stream>>>(yout, nullptr, nullptr, gammab, betab, lengths, 1);
    } else {
      ln_kernel<<<(BBATCH * TT + 255) / 256, 256, 0, stream>>>(yout, bp.l2g, bp.l2b, nullptr, nullptr, lengths, 1);
    }
    float* tmp = xin; xin = yout; yout = tmp;
  }

  gemm(proj_w1, CD, 1, 0, xin, TT, 1, (long)CD * TT,
       proj_b1, nullptr, 0, pbuf, (long)CD * TT, CD, TT, CD, 1, 0);
  gemm(proj_w2, CD, 1, 0, pbuf, TT, 1, (long)CD * TT,
       proj_b2, nullptr, 0, netout, (long)MELD * TT, MELD, TT, CD, 0, 0);

  final_kernel<<<(BBATCH * MELD * TT + 255) / 256, 256, 0, stream>>>(
      mels, noises, noised, netout, times, (float*)d_out);
}
